// EnhancedAttention_69904887710043
// MI455X (gfx1250) — matrix operand order repaired and checked by reading, not compiled, not run on hardware
//
#include <hip/hip_runtime.h>
#include <hip/hip_bf16.h>
#include <math.h>

// Problem constants
#define B_   2
#define S_   2048
#define DM_  1024
#define H_   16
#define D_   64
#define NQKV 3072   // 3*H*D

#define KC    64    // k-elements staged per chunk (GEMM)
#define AKC   64    // keys staged per chunk (attention)
#define BROW  144   // padded LDS row stride in bytes (64*2 = 128 data + 16 pad)

typedef __bf16 bf16;
typedef bf16  v8bf  __attribute__((ext_vector_type(8)));
typedef bf16  v16bf __attribute__((ext_vector_type(16)));
typedef float v8f   __attribute__((ext_vector_type(8)));

static __device__ __forceinline__ v16bf mk16(v8bf lo, v8bf hi) {
  return __builtin_shufflevector(lo, hi, 0,1,2,3,4,5,6,7,8,9,10,11,12,13,14,15);
}
static __device__ __forceinline__ v8bf ld8(const bf16* p) { return *(const v8bf*)p; }

static __device__ __forceinline__ unsigned pk2(float a, float b) {
  union { bf16 h[2]; unsigned u; } x;
  x.h[0] = (bf16)a; x.h[1] = (bf16)b;
  return x.u;
}

static __device__ __forceinline__ v8f wmma_bf16(v16bf a, v16bf b, v8f c) {
  // D = A(16x32 bf16) * B(32x16 bf16) + C(16x16 f32)
  return __builtin_amdgcn_wmma_f32_16x16x32_bf16(false, a, false, b, (short)0, c,
                                                 false, false);
}

// One async 16-byte global->LDS copy per lane (GLOBAL_LOAD_ASYNC_TO_LDS_B128,
// GVS mode: mem = SGPR64 + VGPR32). Tracked by ASYNCcnt.
static __device__ __forceinline__ void async_cp16(unsigned ldsOff, unsigned voff,
                                                  unsigned long long sbase) {
  asm volatile("global_load_async_to_lds_b128 %0, %1, %2"
               :: "v"(ldsOff), "v"(voff), "s"(sbase)
               : "memory");
}
static __device__ __forceinline__ void wait_async0() {
  asm volatile("s_wait_asynccnt 0x0" ::: "memory");
}
static __device__ __forceinline__ void wait_async2() {
  asm volatile("s_wait_asynccnt 0x2" ::: "memory");
}
static __device__ __forceinline__ void wait_async4() {
  asm volatile("s_wait_asynccnt 0x4" ::: "memory");
}

// ---------------------------------------------------------------------------
// Kernel 1: fp32 -> bf16 bulk convert (x)
// ---------------------------------------------------------------------------
__global__ void __launch_bounds__(256) k_cvt_bf16(const float* __restrict__ in,
                                                  bf16* __restrict__ out, int n4) {
  int i = blockIdx.x * blockDim.x + threadIdx.x;
  if (i < n4) {
    float4 v = ((const float4*)in)[i];
    union { bf16 h[4]; uint2 u; } o;
    o.h[0] = (bf16)v.x; o.h[1] = (bf16)v.y; o.h[2] = (bf16)v.z; o.h[3] = (bf16)v.w;
    ((uint2*)out)[i] = o.u;
  }
}

// ---------------------------------------------------------------------------
// Kernel 2: transpose-pack W (K x N fp32) -> Wt (N x K bf16), LDS tile transpose
// ---------------------------------------------------------------------------
__global__ void k_pack_t(const float* __restrict__ W, bf16* __restrict__ Wt,
                         int K, int N) {
  __shared__ float t[16][17];
  int k0 = blockIdx.y * 16, n0 = blockIdx.x * 16;
  int tx = threadIdx.x, ty = threadIdx.y;
  t[ty][tx] = W[(size_t)(k0 + ty) * N + (n0 + tx)];
  __syncthreads();
  Wt[(size_t)(n0 + ty) * K + (k0 + tx)] = (bf16)t[tx][ty];
}

// ---------------------------------------------------------------------------
// Kernel 2b: sinusoidal relative position encoding table pe[S][D] (fp32)
// ---------------------------------------------------------------------------
__global__ void __launch_bounds__(256) k_pe(float* __restrict__ pe) {
  int i = blockIdx.x * blockDim.x + threadIdx.x;   // S_*D_ threads
  int s = i >> 6, d = i & 63;
  int i2 = d & ~1;
  float freq = expf((float)i2 * (-9.210340371976184f / 64.f));  // -ln(1e4)/D
  float ang  = (float)s * freq;
  pe[i] = (d & 1) ? cosf(ang) : sinf(ang);
}

// ---------------------------------------------------------------------------
// Kernel 3: QKV GEMM (4096 x 3072) = Xb @ Wqkv^T-packed + bias.
// Block = 8 waves sharing one 64-column B strip, double-buffered async->LDS.
// Wave computes a 16x64 strip (4 C tiles). Epilogue scatters Q/K/Vt in bf16.
// ---------------------------------------------------------------------------
__global__ void __launch_bounds__(256) k_qkv(const bf16* __restrict__ Xb,
                                             const bf16* __restrict__ Wt,
                                             const float* __restrict__ bias,
                                             const float* __restrict__ pe,
                                             bf16* __restrict__ Qb,
                                             bf16* __restrict__ Kb,
                                             bf16* __restrict__ Vt) {
  __shared__ __align__(16) char ldsB[2][64 * BROW];

  const int lane  = threadIdx.x & 31;
  const int wave  = threadIdx.x >> 5;
  const int nG    = blockIdx.x;              // 0..47  (64-column strip)
  const int mTile = blockIdx.y * 8 + wave;   // 0..255
  const int hi    = lane >> 4;
  const int col   = lane & 15;

  const bf16* arow = Xb + (size_t)(mTile * 16 + col) * DM_;
  const unsigned ldsBase = (unsigned)(uintptr_t)&ldsB[0][0];
  const unsigned long long wtBase =
      (unsigned long long)(uintptr_t)Wt + (unsigned long long)nG * 64 * (DM_ * 2);
  const int id0 = wave * 64 + lane;          // this thread's copy ids (+32)

  // prefetch chunk 0
  #pragma unroll
  for (int it = 0; it < 2; it++) {
    int id = id0 + it * 32;
    async_cp16(ldsBase + (unsigned)((id >> 3) * BROW + (id & 7) * 16),
               (unsigned)((id >> 3) * (DM_ * 2) + (id & 7) * 16), wtBase);
  }

  v8f acc[4] = {};
  const int NCH = DM_ / KC;                  // 16 chunks
  for (int ch = 0; ch < NCH; ch++) {
    if (ch + 1 < NCH) {
      unsigned bufo = (unsigned)(((ch + 1) & 1) * (64 * BROW));
      unsigned long long sb = wtBase + (unsigned long long)(ch + 1) * KC * 2;
      #pragma unroll
      for (int it = 0; it < 2; it++) {
        int id = id0 + it * 32;
        async_cp16(ldsBase + bufo + (unsigned)((id >> 3) * BROW + (id & 7) * 16),
                   (unsigned)((id >> 3) * (DM_ * 2) + (id & 7) * 16), sb);
      }
      wait_async2();                         // current chunk's copies landed
    } else {
      wait_async0();
    }
    __syncthreads();                         // all waves' copies visible

    const char* bbuf = &ldsB[ch & 1][0];
    #pragma unroll
    for (int ks = 0; ks < 2; ks++) {
      int k0 = ch * KC + ks * 32;
      v16bf af = mk16(ld8(arow + k0 + hi * 16), ld8(arow + k0 + hi * 16 + 8));
      #pragma unroll
      for (int j = 0; j < 4; j++) {
        const bf16* bp =
            (const bf16*)(bbuf + (j * 16 + col) * BROW + ks * 64 + hi * 32);
        acc[j] = wmma_bf16(af, mk16(ld8(bp), ld8(bp + 8)), acc[j]);
      }
    }
    __syncthreads();                         // buffer reusable next iteration
  }

  const float QSCALE = 0.125f * 1.44269504088896340736f;  // 1/sqrt(64)*log2(e)
  #pragma unroll
  for (int j = 0; j < 4; j++) {
    int gc = nG * 64 + j * 16 + col;         // column in [0, 3072)
    float bv = bias[gc];
    int sel = gc >> 10;                       // 0=Q 1=K 2=V (uniform per wave)
    int hd  = gc & 1023;
    int h   = hd >> 6;
    int d   = hd & 63;
    #pragma unroll
    for (int r = 0; r < 8; r++) {
      int gr = mTile * 16 + r + 8 * hi;       // row in [0, 4096)
      int bb = gr >> 11;
      int s  = gr & 2047;
      float v = acc[j][r] + bv;
      if (sel == 0) {
        Qb[(((size_t)(bb * H_ + h)) * S_ + s) * D_ + d] = (bf16)(v * QSCALE);
      } else if (sel == 1) {
        Kb[(((size_t)(bb * H_ + h)) * S_ + s) * D_ + d] =
            (bf16)(v + pe[(size_t)s * D_ + d]);
      } else {
        Vt[(((size_t)(bb * H_ + h)) * D_ + d) * S_ + s] = (bf16)v;  // transposed
      }
    }
  }
}

// ---------------------------------------------------------------------------
// Kernel 4: flash attention. Block = 8 waves sharing one (b,h); wave = 16
// queries. K/V chunks (64 keys) double-buffered into LDS with async copies.
// S^T = K @ Q^T (softmax = in-lane + one xor-16 shuffle),
// O^T = V^T @ P^T (P^T B-fragment packed in-register straight from the S^T accumulators).
// ---------------------------------------------------------------------------
__global__ void __launch_bounds__(256) k_attn(const bf16* __restrict__ Qb,
                                              const bf16* __restrict__ Kb,
                                              const bf16* __restrict__ Vt,
                                              bf16* __restrict__ Ob) {
  __shared__ __align__(16) char ldsK[2][64 * BROW];  // 64 key-rows x 64 d
  __shared__ __align__(16) char ldsV[2][64 * BROW];  // 64 d-rows x 64 keys

  const int lane = threadIdx.x & 31;
  const int wave = threadIdx.x >> 5;
  const int bh   = blockIdx.x >> 4;                 // b*16 + h
  const int q0   = (blockIdx.x & 15) * 128 + wave * 16;
  const int hi   = lane >> 4;
  const int col  = lane & 15;

  // Q as B-operand of S^T = K @ Q^T : table order (8h+s, 16+8h+s) per d-chunk
  const bf16* qrow = Qb + ((size_t)bh * S_ + (q0 + col)) * D_;
  v16bf qf0 = mk16(ld8(qrow + hi * 8),      ld8(qrow + 16 + hi * 8));
  v16bf qf1 = mk16(ld8(qrow + 32 + hi * 8), ld8(qrow + 48 + hi * 8));

  const unsigned ldsKb = (unsigned)(uintptr_t)&ldsK[0][0];
  const unsigned ldsVb = (unsigned)(uintptr_t)&ldsV[0][0];
  const unsigned long long kBase =
      (unsigned long long)(uintptr_t)Kb + (unsigned long long)bh * S_ * D_ * 2;
  const unsigned long long vBase =
      (unsigned long long)(uintptr_t)Vt + (unsigned long long)bh * D_ * S_ * 2;
  const int id0 = wave * 64 + lane;

  // prefetch chunk 0 (keys 0..63): K rows contiguous, Vt rows strided by S
  #pragma unroll
  for (int it = 0; it < 2; it++) {
    int id = id0 + it * 32;
    unsigned lo = (unsigned)((id >> 3) * BROW + (id & 7) * 16);
    async_cp16(ldsKb + lo, (unsigned)((id >> 3) * (D_ * 2) + (id & 7) * 16), kBase);
    async_cp16(ldsVb + lo, (unsigned)((id >> 3) * (S_ * 2) + (id & 7) * 16), vBase);
  }

  float m = -INFINITY, l = 0.f;
  v8f o0 = {}, o1 = {}, o2 = {}, o3 = {};

  const int NCH = S_ / AKC;                         // 32 chunks
  for (int ch = 0; ch < NCH; ch++) {
    if (ch + 1 < NCH) {
      unsigned bufo = (unsigned)(((ch + 1) & 1) * (64 * BROW));
      int kc = (ch + 1) * AKC;
      unsigned long long ksb = kBase + (unsigned long long)kc * (D_ * 2);
      unsigned long long vsb = vBase + (unsigned long long)kc * 2;
      #pragma unroll
      for (int it = 0; it < 2; it++) {
        int id = id0 + it * 32;
        unsigned lo = bufo + (unsigned)((id >> 3) * BROW + (id & 7) * 16);
        async_cp16(ldsKb + lo, (unsigned)((id >> 3) * (D_ * 2) + (id & 7) * 16), ksb);
        async_cp16(ldsVb + lo, (unsigned)((id >> 3) * (S_ * 2) + (id & 7) * 16), vsb);
      }
      wait_async4();
    } else {
      wait_async0();
    }
    __syncthreads();

    const char* kbuf = &ldsK[ch & 1][0];
    const char* vbuf = &ldsV[ch & 1][0];

    #pragma unroll
    for (int hf = 0; hf < 2; hf++) {                // two 32-key halves
      // --- scores^T tiles (M=key, N=query) from LDS K ---
      v8f st[2];
      #pragma unroll
      for (int t = 0; t < 2; t++) {
        const char* kr = kbuf + (hf * 32 + t * 16 + col) * BROW;
        v16bf ka0 = mk16(ld8((const bf16*)(kr + hi * 16)),
                         ld8((const bf16*)(kr + 32 + hi * 16)));
        v16bf ka1 = mk16(ld8((const bf16*)(kr + 64 + hi * 16)),
                         ld8((const bf16*)(kr + 96 + hi * 16)));
        v8f s = {};
        s = wmma_bf16(ka0, qf0, s);
        s = wmma_bf16(ka1, qf1, s);
        st[t] = s;
      }

      // --- online softmax (log2 domain; Q pre-scaled) ---
      float cm = st[0][0];
      #pragma unroll
      for (int e = 1; e < 8; e++) cm = fmaxf(cm, st[0][e]);
      #pragma unroll
      for (int e = 0; e < 8; e++) cm = fmaxf(cm, st[1][e]);
      cm = fmaxf(cm, __shfl_xor(cm, 16, 32));
      float nm = fmaxf(m, cm);
      float rs = exp2f(m - nm);
      m = nm;
      l *= rs;
      #pragma unroll
      for (int e = 0; e < 8; e++) {
        st[0][e] = exp2f(st[0][e] - nm);
        st[1][e] = exp2f(st[1][e] - nm);
        l += st[0][e] + st[1][e];
      }
      o0 = o0 * rs; o1 = o1 * rs; o2 = o2 * rs; o3 = o3 * rs;

      // --- assemble P^T B-fragment (32 keys x 16 queries) in-register ---
      unsigned pa[4], pb[4], oa[4], ob[4];
      #pragma unroll
      for (int v = 0; v < 4; v++) {
        pa[v] = pk2(st[0][2 * v], st[0][2 * v + 1]);
        pb[v] = pk2(st[1][2 * v], st[1][2 * v + 1]);
      }
      #pragma unroll
      for (int v = 0; v < 4; v++) {
        oa[v] = (unsigned)__shfl_xor((int)pa[v], 16, 32);
        ob[v] = (unsigned)__shfl_xor((int)pb[v], 16, 32);
      }
      union { v16bf v; unsigned u[8]; } pf;
      #pragma unroll
      for (int v = 0; v < 4; v++) {
        pf.u[v]     = pa[v];    // K= 0..7  | 8..15
        pf.u[v + 4] = pb[v];    // K= 16..23 | 24..31
      }

      // --- O^T += V^T @ P^T : 4 d-tiles, A-fragments from LDS V ---
      const char* v0p = vbuf + (0  + col) * BROW + hf * 64;
      o0 = wmma_bf16(mk16(ld8((const bf16*)(v0p + hi * 16)),
                          ld8((const bf16*)(v0p + 32 + hi * 16))), pf.v, o0);
      const char* v1p = vbuf + (16 + col) * BROW + hf * 64;
      o1 = wmma_bf16(mk16(ld8((const bf16*)(v1p + hi * 16)),
                          ld8((const bf16*)(v1p + 32 + hi * 16))), pf.v, o1);
      const char* v2p = vbuf + (32 + col) * BROW + hf * 64;
      o2 = wmma_bf16(mk16(ld8((const bf16*)(v2p + hi * 16)),
                          ld8((const bf16*)(v2p + 32 + hi * 16))), pf.v, o2);
      const char* v3p = vbuf + (48 + col) * BROW + hf * 64;
      o3 = wmma_bf16(mk16(ld8((const bf16*)(v3p + hi * 16)),
                          ld8((const bf16*)(v3p + 32 + hi * 16))), pf.v, o3);
    }
    __syncthreads();
  }

  float lt  = l + __shfl_xor(l, 16, 32);
  float inv = 1.f / lt;

  // Store O (row = b*S+q, col = h*64+d) as bf16, 16B per lane per d-tile
  bf16* orow = Ob + ((size_t)(bh >> 4) * S_ + q0 + col) * DM_ + (size_t)(bh & 15) * D_;
  v8f oo[4] = {o0, o1, o2, o3};
  #pragma unroll
  for (int dt = 0; dt < 4; dt++) {
    unsigned u[4];
    #pragma unroll
    for (int v = 0; v < 4; v++)
      u[v] = pk2(oo[dt][2 * v] * inv, oo[dt][2 * v + 1] * inv);
    *(uint4*)(orow + dt * 16 + hi * 8) = *(const uint4*)u;
  }
}

// ---------------------------------------------------------------------------
// Kernel 5: output projection (4096 x 1024) = Ob @ Wout + bias -> fp32 d_out.
// Same async->LDS double-buffered B staging as k_qkv.
// ---------------------------------------------------------------------------
__global__ void __launch_bounds__(256) k_out(const bf16* __restrict__ Ab,
                                             const bf16* __restrict__ Wt,
                                             const float* __restrict__ bias,
                                             float* __restrict__ out) {
  __shared__ __align__(16) char ldsB[2][64 * BROW];

  const int lane  = threadIdx.x & 31;
  const int wave  = threadIdx.x >> 5;
  const int nG    = blockIdx.x;              // 0..15
  const int mTile = blockIdx.y * 8 + wave;   // 0..255
  const int hi    = lane >> 4;
  const int col   = lane & 15;

  const bf16* arow = Ab + (size_t)(mTile * 16 + col) * DM_;
  const unsigned ldsBase = (unsigned)(uintptr_t)&ldsB[0][0];
  const unsigned long long wtBase =
      (unsigned long long)(uintptr_t)Wt + (unsigned long long)nG * 64 * (DM_ * 2);
  const int id0 = wave * 64 + lane;

  #pragma unroll
  for (int it = 0; it < 2; it++) {
    int id = id0 + it * 32;
    async_cp16(ldsBase + (unsigned)((id >> 3) * BROW + (id & 7) * 16),
               (unsigned)((id >> 3) * (DM_ * 2) + (id & 7) * 16), wtBase);
  }

  v8f acc[4] = {};
  const int NCH = DM_ / KC;
  for (int ch = 0; ch < NCH; ch++) {
    if (ch + 1 < NCH) {
      unsigned bufo = (unsigned)(((ch + 1) & 1) * (64 * BROW));
      unsigned long long sb = wtBase + (unsigned long long)(ch + 1) * KC * 2;
      #pragma unroll
      for (int it = 0; it < 2; it++) {
        int id = id0 + it * 32;
        async_cp16(ldsBase + bufo + (unsigned)((id >> 3) * BROW + (id & 7) * 16),
                   (unsigned)((id >> 3) * (DM_ * 2) + (id & 7) * 16), sb);
      }
      wait_async2();
    } else {
      wait_async0();
    }
    __syncthreads();

    const char* bbuf = &ldsB[ch & 1][0];
    #pragma unroll
    for (int ks = 0; ks < 2; ks++) {
      int k0 = ch * KC + ks * 32;
      v16bf af = mk16(ld8(arow + k0 + hi * 16), ld8(arow + k0 + hi * 16 + 8));
      #pragma unroll
      for (int j = 0; j < 4; j++) {
        const bf16* bp =
            (const bf16*)(bbuf + (j * 16 + col) * BROW + ks * 64 + hi * 32);
        acc[j] = wmma_bf16(af, mk16(ld8(bp), ld8(bp + 8)), acc[j]);
      }
    }
    __syncthreads();
  }

  #pragma unroll
  for (int j = 0; j < 4; j++) {
    int gc = nG * 64 + j * 16 + col;
    float bv = bias[gc];
    #pragma unroll
    for (int r = 0; r < 8; r++) {
      int gr = mTile * 16 + r + 8 * hi;
      out[(size_t)gr * DM_ + gc] = acc[j][r] + bv;
    }
  }
}

// ---------------------------------------------------------------------------
extern "C" void kernel_launch(void* const* d_in, const int* in_sizes, int n_in,
                              void* d_out, int out_size, void* d_ws, size_t ws_size,
                              hipStream_t stream) {
  const float* x    = (const float*)d_in[0];
  const float* Wqkv = (const float*)d_in[1];
  const float* bqkv = (const float*)d_in[2];
  const float* Wout = (const float*)d_in[3];
  const float* bout = (const float*)d_in[4];
  float* out = (float*)d_out;

  char* ws = (char*)d_ws;
  size_t off = 0;
  auto alloc = [&](size_t bytes) -> void* {
    void* p = ws + off;
    off = (off + bytes + 255) & ~(size_t)255;
    return p;
  };
  bf16*  xb  = (bf16*)alloc((size_t)4096 * 1024 * 2);      // x in bf16
  bf16*  wtq = (bf16*)alloc((size_t)3072 * 1024 * 2);      // W_qkv^T bf16
  bf16*  wto = (bf16*)alloc((size_t)1024 * 1024 * 2);      // W_out^T bf16
  bf16*  Qb  = (bf16*)alloc((size_t)B_ * H_ * S_ * D_ * 2); // [B][H][S][D]
  bf16*  Kb  = (bf16*)alloc((size_t)B_ * H_ * S_ * D_ * 2); // [B][H][S][D]
  bf16*  Vt  = (bf16*)alloc((size_t)B_ * H_ * D_ * S_ * 2); // [B][H][D][S]
  bf16*  Ob  = (bf16*)alloc((size_t)4096 * 1024 * 2);      // attention out bf16
  float* pe  = (float*)alloc((size_t)S_ * D_ * 4);         // pos-enc table

  hipLaunchKernelGGL(k_cvt_bf16, dim3(4096), dim3(256), 0, stream,
                     x, xb, 4096 * 1024 / 4);
  hipLaunchKernelGGL(k_pack_t, dim3(NQKV / 16, DM_ / 16), dim3(16, 16), 0, stream,
                     Wqkv, wtq, DM_, NQKV);
  hipLaunchKernelGGL(k_pack_t, dim3(DM_ / 16, DM_ / 16), dim3(16, 16), 0, stream,
                     Wout, wto, DM_, DM_);
  hipLaunchKernelGGL(k_pe, dim3(S_ * D_ / 256), dim3(256), 0, stream, pe);
  hipLaunchKernelGGL(k_qkv, dim3(NQKV / 64, 4096 / 128), dim3(256), 0, stream,
                     xb, wtq, bqkv, pe, Qb, Kb, Vt);
  hipLaunchKernelGGL(k_attn, dim3((B_ * H_) * (S_ / 128)), dim3(256), 0, stream,
                     Qb, Kb, Vt, Ob);
  hipLaunchKernelGGL(k_out, dim3(DM_ / 64, 4096 / 128), dim3(256), 0, stream,
                     Ob, wto, bout, out);
}
